// Attention_81449759801741
// MI455X (gfx1250) — matrix operand order repaired and checked by reading, not compiled, not run on hardware
//
#include <hip/hip_runtime.h>
#include <hip/hip_bf16.h>
#include <math.h>

// ---------------------------------------------------------------------------
// MI455X (gfx1250) attention pipeline, all matmuls via v_wmma_f32_16x16x32_bf16.
// Compute-bound (~275 GFLOP, ~900 FLOP/byte) -> bf16 WMMA, f32 accum.
// B operands pre-transposed to Bt[n][k]; LDS tiles staged with
// GLOBAL_LOAD_ASYNC_TO_LDS_B128 (ASYNCcnt), double-buffered; fragment loads are
// contiguous ds_load_b128 pairs matching ISA 7.12.2 layouts.
// Wave tile 64x64 -> 16 WMMA per K-step per wave (1.0 wmma per ds_load_b128).
// ---------------------------------------------------------------------------

typedef __attribute__((ext_vector_type(16))) __bf16 v16bf;
typedef __attribute__((ext_vector_type(8)))  float  v8f;

#define LDS_PITCH 40   // 32 bf16 + 8 pad (80 B rows: keeps 16B align, rotates banks)

// -------- CDNA5 async global->LDS staging (ASYNCcnt-tracked DMA path) -------
template<int OFF>
__device__ __forceinline__ void async_copy_b128(const void* gsrc, unsigned lds_off)
{
    // offset immediate applies to BOTH the LDS destination and global source
    // (ISA 10.x) -> contiguous row copy.
    asm volatile("global_load_async_to_lds_b128 %0, %1, off offset:%2"
                 :: "v"(lds_off), "v"((unsigned long long)gsrc), "i"(OFF)
                 : "memory");
}
__device__ __forceinline__ void wait_async0()
{
    asm volatile("s_wait_asynccnt 0x0" ::: "memory");
}

// ----------------------------- f32 -> bf16 convert -------------------------
__global__ __launch_bounds__(256) void f32_to_bf16_kernel(
    const float* __restrict__ in, __bf16* __restrict__ out, long long n)
{
    long long i = (long long)blockIdx.x * 256 + threadIdx.x;
    if (i < n) out[i] = (__bf16)in[i];
}

// in[R][C] f32  ->  out[C][R] bf16   (one-time weight transpose)
__global__ __launch_bounds__(256) void f32_to_bf16_transpose_kernel(
    const float* __restrict__ in, __bf16* __restrict__ out, int R, int C)
{
    long long i = (long long)blockIdx.x * 256 + threadIdx.x;
    if (i < (long long)R * C) {
        int r = (int)(i / C), c = (int)(i % C);
        out[(long long)c * R + r] = (__bf16)in[i];
    }
}

// ----------------------------- WMMA GEMM -----------------------------------
// C[M,N] = A[M,K] * B + bias(+mask).  A row-major (lda=K).
// Bt is the TRANSPOSED B: Bt[n][k], leading dim ldB.
// Block tile 128(M) x 256(N), 256 threads = 8 wave32s as 2(M) x 4(N);
// each wave: 64x64 = 4x4 frags of 16x16, K-step 32, async double-buffered LDS.
// OUT_MODE: 0 = bf16 row-major, 1 = bf16 TRANSPOSED (C^T[n][m], ldC=M),
//           2 = f32 row-major.
template<int OUT_MODE, bool APPLY_MASK>
__global__ __launch_bounds__(256) void gemm_wmma_bf16(
    const __bf16* __restrict__ A, const __bf16* __restrict__ Bt,
    const float* __restrict__ bias, const int* __restrict__ mask,
    __bf16* __restrict__ Cb, float* __restrict__ Cf,
    int M, int N, int K, int ldB, int ldC,
    long long sA, long long sB, long long sC, int maskStride)
{
    __shared__ __align__(16) __bf16 As[2][128 * LDS_PITCH];   // [m][k]
    __shared__ __align__(16) __bf16 Bs[2][256 * LDS_PITCH];   // [n][k]

    const int t    = threadIdx.x;
    const int lane = t & 31;
    const int w    = t >> 5;
    const int half = lane >> 4;      // lane group 0/1
    const int l16  = lane & 15;
    const int wm   = (w & 1) * 64;   // wave M offset in block tile (2 waves)
    const int wn   = (w >> 1) * 64;  // wave N offset in block tile (4 waves)

    const long long z = blockIdx.z;
    const int m0 = blockIdx.y * 128;
    const int n0 = blockIdx.x * 256;

    // staging: per thread, A = 16 contiguous bf16 (2 async b128),
    //          B = one full 32-elem row (4 async b128)
    const int arow = t >> 1;            // 0..127
    const int skc  = (t & 1) * 16;      // 0 or 16
    const __bf16* aSrcBase = A  + z * sA + (long long)(m0 + arow) * K   + skc;
    const __bf16* bSrcBase = Bt + z * sB + (long long)(n0 + t)    * ldB;

    const unsigned aOffBuf[2] = {
        (unsigned)(size_t)&As[0][arow * LDS_PITCH + skc],
        (unsigned)(size_t)&As[1][arow * LDS_PITCH + skc] };
    const unsigned bOffBuf[2] = {
        (unsigned)(size_t)&Bs[0][t * LDS_PITCH],
        (unsigned)(size_t)&Bs[1][t * LDS_PITCH] };

    v8f acc[4][4];
    #pragma unroll
    for (int mi = 0; mi < 4; ++mi)
        #pragma unroll
        for (int ni = 0; ni < 4; ++ni)
            #pragma unroll
            for (int r = 0; r < 8; ++r) acc[mi][ni][r] = 0.0f;

    // prologue: async-stage k0 = 0 into buffer 0
    {
        const __bf16* as = aSrcBase;
        const __bf16* bs = bSrcBase;
        async_copy_b128<0 >(as, aOffBuf[0]);
        async_copy_b128<16>(as, aOffBuf[0]);
        async_copy_b128<0 >(bs, bOffBuf[0]);
        async_copy_b128<16>(bs, bOffBuf[0]);
        async_copy_b128<32>(bs, bOffBuf[0]);
        async_copy_b128<48>(bs, bOffBuf[0]);
    }
    wait_async0();
    __syncthreads();

    union Frag { v16bf v; uint4 u[2]; };

    for (int k0 = 0; k0 < K; k0 += 32) {
        const int cur   = (k0 >> 5) & 1;
        const bool more = (k0 + 32) < K;

        // issue async DMA for next tile into the other buffer (no VGPR data)
        if (more) {
            const int nxt = cur ^ 1;
            const __bf16* as = aSrcBase + k0 + 32;
            const __bf16* bs = bSrcBase + k0 + 32;
            async_copy_b128<0 >(as, aOffBuf[nxt]);
            async_copy_b128<16>(as, aOffBuf[nxt]);
            async_copy_b128<0 >(bs, bOffBuf[nxt]);
            async_copy_b128<16>(bs, bOffBuf[nxt]);
            async_copy_b128<32>(bs, bOffBuf[nxt]);
            async_copy_b128<48>(bs, bOffBuf[nxt]);
        }

        // ---- fragment loads: pure ds_load_b128 pairs (ISA 7.12.2 layouts) ----
        Frag a[4], b[4];
        #pragma unroll
        for (int mi = 0; mi < 4; ++mi) {
            const int base = (wm + mi * 16 + l16) * LDS_PITCH + 8 * half;
            a[mi].u[0] = *reinterpret_cast<const uint4*>(&As[cur][base]);       // k=8h..8h+7
            a[mi].u[1] = *reinterpret_cast<const uint4*>(&As[cur][base + 16]);  // k=16+8h..
        }
        #pragma unroll
        for (int ni = 0; ni < 4; ++ni) {
            const int base = (wn + ni * 16 + l16) * LDS_PITCH + 8 * half;
            b[ni].u[0] = *reinterpret_cast<const uint4*>(&Bs[cur][base]);       // k=8h..8h+7
            b[ni].u[1] = *reinterpret_cast<const uint4*>(&Bs[cur][base + 16]);  // k=16+8h..
        }

        #pragma unroll
        for (int mi = 0; mi < 4; ++mi)
            #pragma unroll
            for (int ni = 0; ni < 4; ++ni)
                acc[mi][ni] = __builtin_amdgcn_wmma_f32_16x16x32_bf16(
                    false, a[mi].v, false, b[ni].v, (short)0, acc[mi][ni],
                    false, false);

        if (more) wait_async0();   // own async DMA done before barrier
        __syncthreads();
    }

    // ---- epilogue (C/D layout: row = r + 8*half, col = lane&15) ----
    #pragma unroll
    for (int mi = 0; mi < 4; ++mi) {
        #pragma unroll
        for (int ni = 0; ni < 4; ++ni) {
            const int col  = n0 + wn + ni * 16 + l16;
            const int rowb = m0 + wm + mi * 16 + (half << 3);
            const float bvv = bias ? bias[col] : 0.0f;
            bool masked = false;
            if (APPLY_MASK) masked = (mask[z * maskStride + col] == 0);

            if (OUT_MODE == 1) {
                // transposed bf16 store: 8 consecutive rows -> one b128
                union { uint4 u; __bf16 h[8]; } pk;
                #pragma unroll
                for (int r = 0; r < 8; ++r)
                    pk.h[r] = (__bf16)(acc[mi][ni][r] + bvv);
                *reinterpret_cast<uint4*>(
                    &Cb[z * sC + (long long)col * ldC + rowb]) = pk.u;
            } else {
                #pragma unroll
                for (int r = 0; r < 8; ++r) {
                    float vv = acc[mi][ni][r] + bvv;
                    if (APPLY_MASK && masked) vv = -10000.0f;
                    const long long idx = z * sC + (long long)(rowb + r) * ldC + col;
                    if (OUT_MODE == 2) Cf[idx] = vv;
                    else               Cb[idx] = (__bf16)vv;
                }
            }
        }
    }
}

// ----------------------------- row softmax (f32 -> bf16) -------------------
// One block (256 threads) per row of 2048 columns (8 per thread).
__global__ __launch_bounds__(256) void softmax_rows_kernel(
    const float* __restrict__ S, __bf16* __restrict__ P, int ncols)
{
    const int t = threadIdx.x;
    const long long row = blockIdx.x;
    const float* s = S + row * ncols;
    __bf16* p = P + row * ncols;

    float v[8];
    float m = -3.402823e38f;
    #pragma unroll
    for (int j = 0; j < 8; ++j) {
        v[j] = s[t + j * 256];
        m = fmaxf(m, v[j]);
    }
    #pragma unroll
    for (int off = 16; off > 0; off >>= 1)
        m = fmaxf(m, __shfl_xor(m, off, 32));

    __shared__ float red[8];
    if ((t & 31) == 0) red[t >> 5] = m;
    __syncthreads();
    float bm = red[0];
    #pragma unroll
    for (int i = 1; i < 8; ++i) bm = fmaxf(bm, red[i]);
    __syncthreads();

    float sum = 0.0f;
    #pragma unroll
    for (int j = 0; j < 8; ++j) {
        v[j] = __expf(v[j] - bm);
        sum += v[j];
    }
    #pragma unroll
    for (int off = 16; off > 0; off >>= 1)
        sum += __shfl_xor(sum, off, 32);
    if ((t & 31) == 0) red[t >> 5] = sum;
    __syncthreads();
    float bs = 0.0f;
    #pragma unroll
    for (int i = 0; i < 8; ++i) bs += red[i];
    const float inv = 1.0f / bs;

    #pragma unroll
    for (int j = 0; j < 8; ++j)
        p[t + j * 256] = (__bf16)(v[j] * inv);
}

// ----------------------------- launcher ------------------------------------
extern "C" void kernel_launch(void* const* d_in, const int* in_sizes, int n_in,
                              void* d_out, int out_size, void* d_ws, size_t ws_size,
                              hipStream_t stream)
{
    const float* q    = (const float*)d_in[0];
    const float* k    = (const float*)d_in[1];
    const int*   mask = (const int*)  d_in[2];
    const float* Wq   = (const float*)d_in[3];
    const float* bq   = (const float*)d_in[4];
    const float* Wk   = (const float*)d_in[5];
    const float* bk   = (const float*)d_in[6];
    const float* Wv   = (const float*)d_in[7];
    const float* bv   = (const float*)d_in[8];
    const float* Wo   = (const float*)d_in[9];
    const float* bo   = (const float*)d_in[10];
    float* out = (float*)d_out;

    constexpr int  Bn = 8, LQ = 2048, LK = 2048, H = 1024;
    constexpr int  MF = Bn * LQ;                       // folded batch M = 16384
    constexpr long long NQ = (long long)MF * H;        // 16,777,216
    constexpr long long NW = (long long)H * H;         // 1,048,576

    // ---- workspace layout (bytes), aliased, peak ~296 MB ----
    char* ws = (char*)d_ws;
    __bf16* qbf  = (__bf16*)(ws + 0);                  // 32 MB (dead after qh)
    __bf16* kbf  = (__bf16*)(ws + 33554432);           // 32 MB (dead after kh)
    __bf16* wqT  = (__bf16*)(ws + 67108864);           // 2 MB  (Wq^T)
    __bf16* wkT  = (__bf16*)(ws + 69206016);           // 2 MB
    __bf16* wvT  = (__bf16*)(ws + 71303168);           // 2 MB
    __bf16* woT  = (__bf16*)(ws + 73400320);           // 2 MB
    __bf16* qh   = (__bf16*)(ws + 75497472);           // 32 MB
    __bf16* kh   = (__bf16*)(ws + 109051904);          // 32 MB
    __bf16* vhT  = (__bf16*)(ws + 142606336);          // 32 MB: vh^T [H][B*LK]
    float*  sc   = (float*) (ws + 176160768);          // 128 MB (dead after softmax)
    __bf16* probs= (__bf16*)(ws + 0);                  // 64 MB, reuses qbf+kbf
    __bf16* attn = (__bf16*)(ws + 176160768);          // 32 MB, reuses scores

    // ---- 1) conversions: q,k linear; weights transposed to Bt[n][k] ----
    f32_to_bf16_kernel<<<(unsigned)((NQ + 255) / 256), 256, 0, stream>>>(q, qbf, NQ);
    f32_to_bf16_kernel<<<(unsigned)((NQ + 255) / 256), 256, 0, stream>>>(k, kbf, NQ);
    f32_to_bf16_transpose_kernel<<<(unsigned)((NW + 255) / 256), 256, 0, stream>>>(Wq, wqT, H, H);
    f32_to_bf16_transpose_kernel<<<(unsigned)((NW + 255) / 256), 256, 0, stream>>>(Wk, wkT, H, H);
    f32_to_bf16_transpose_kernel<<<(unsigned)((NW + 255) / 256), 256, 0, stream>>>(Wv, wvT, H, H);
    f32_to_bf16_transpose_kernel<<<(unsigned)((NW + 255) / 256), 256, 0, stream>>>(Wo, woT, H, H);

    // ---- 2) qh = q @ Wq + bq  (folded batch) ----
    {
        dim3 g(H / 256, MF / 128, 1);
        gemm_wmma_bf16<0, false><<<g, 256, 0, stream>>>(
            qbf, wqT, bq, nullptr, qh, nullptr,
            MF, H, H, /*ldB=*/H, /*ldC=*/H, 0, 0, 0, 0);
    }
    // ---- 3) kh = k @ Wk + bk ----
    {
        dim3 g(H / 256, MF / 128, 1);
        gemm_wmma_bf16<0, false><<<g, 256, 0, stream>>>(
            kbf, wkT, bk, nullptr, kh, nullptr,
            MF, H, H, H, H, 0, 0, 0, 0);
    }
    // ---- 4) vh^T = (kh @ Wv + bv)^T  (transposed epilogue store) ----
    {
        dim3 g(H / 256, MF / 128, 1);
        gemm_wmma_bf16<1, false><<<g, 256, 0, stream>>>(
            kh, wvT, bv, nullptr, vhT, nullptr,
            MF, H, H, /*ldB=*/H, /*ldC=*/MF, 0, 0, 0, 0);
    }
    // ---- 5) scores = qh @ kh^T, masked -10000 (batched, f32) ----
    // kh is already Bt[n=key][k=h] form.
    {
        dim3 g(LK / 256, LQ / 128, Bn);
        gemm_wmma_bf16<2, true><<<g, 256, 0, stream>>>(
            qh, kh, nullptr, mask, nullptr, sc,
            LQ, LK, H, /*ldB=*/H, /*ldC=*/LK,
            (long long)LQ * H, (long long)LK * H, (long long)LQ * LK, LK);
    }
    // ---- 6) probs = softmax(scores) -> bf16 ----
    softmax_rows_kernel<<<Bn * LQ, 256, 0, stream>>>(sc, probs, LK);

    // ---- 7) attn = probs @ vh (batched).  Bt = vh^T[n=h][B*LK], sB = LK ----
    {
        dim3 g(H / 256, LQ / 128, Bn);
        gemm_wmma_bf16<0, false><<<g, 256, 0, stream>>>(
            probs, vhT, nullptr, nullptr, attn, nullptr,
            LQ, H, LK, /*ldB=*/MF, /*ldC=*/H,
            (long long)LQ * LK, (long long)LK, (long long)LQ * H, 0);
    }
    // ---- 8) out = attn @ Wo + bo (folded batch, f32 to d_out) ----
    {
        dim3 g(H / 256, MF / 128, 1);
        gemm_wmma_bf16<2, false><<<g, 256, 0, stream>>>(
            attn, woT, bo, nullptr, nullptr, out,
            MF, H, H, H, H, 0, 0, 0, 0);
    }
    (void)in_sizes; (void)n_in; (void)out_size; (void)ws_size;
}
